// CrossAttentionFusion_74010876444859
// MI455X (gfx1250) — matrix operand order repaired and checked by reading, not compiled, not run on hardware
//
#include <hip/hip_runtime.h>
#include <hip/hip_bf16.h>
#include <math.h>

typedef __attribute__((ext_vector_type(16))) _Float16 v16h;
typedef __attribute__((ext_vector_type(8)))  _Float16 v8h;
typedef __attribute__((ext_vector_type(8)))  float    v8f;

#define MT   64          // rows per workgroup
#define NTH  512         // 16 waves (wave32)
#define LDA  264         // f16 stride for 256-col LDS buffers (bank-friendly)
#define LDH1 1032        // f16 stride for 1024-col FFN hidden buffer
#define BUFB (MT*LDA*2)  // 33792 bytes per activation buffer

#define OFF_N24  0
#define OFF_N77  (1*BUFB)
#define OFF_NX   (2*BUFB)
#define OFF_P0   (3*BUFB)
#define OFF_P1   (4*BUFB)
#define OFF_P2   (5*BUFB)
#define OFF_P3   (6*BUFB)
#define OFF_D    (7*BUFB)             // f32 [64][192] diff-scores
#define OFF_BPAN (OFF_D + MT*192*4)   // 2 x 16KB weight panels (double buffer)
#define SMEM_BYTES (OFF_BPAN + 32768) // 318464 bytes (<320KB WGP LDS)
#define OFF_F32  OFF_P1               // fused f32 [64][256] (reuses P1+P2)

// ws (f16, WMMA-tiled) weight offsets, in elements
#define WQ24 0
#define WK24 65536
#define WV24 131072
#define WQ77 196608
#define WK77 262144
#define WV77 327680
#define WQX  393216
#define WKX  458752
#define WVX  524288
#define WOUT 589824
#define WFF1 655360
#define WFF2 917504

__device__ __forceinline__ v16h pack16(v8h lo, v8h hi) {
  v16h r;
#pragma unroll
  for (int i = 0; i < 8; ++i) { r[i] = lo[i]; r[i + 8] = hi[i]; }
  return r;
}

// Low 32 bits of a generic shared-memory pointer == LDS byte address
// (FLAT->LDS aperture truncation semantics, ISA 10.2).
__device__ __forceinline__ unsigned lds_addr32(const void* p) {
  return (unsigned)(unsigned long long)p;
}

// Async copy of 32 contiguous bytes per lane: global -> LDS (ASYNCcnt-tracked).
// INST_OFFSET is added to BOTH the LDS and global addresses (ISA 10.7/15.18),
// so one address pair serves both 16B halves.
__device__ __forceinline__ void panel_async_copy32(unsigned ldsAddr, const _Float16* g) {
  asm volatile(
      "global_load_async_to_lds_b128 %0, %1, off\n\t"
      "global_load_async_to_lds_b128 %0, %1, off offset:16"
      :: "v"(ldsAddr), "v"(g)
      : "memory");
}

__device__ __forceinline__ void wait_async0() {
  asm volatile("s_wait_asynccnt 0x0" ::: "memory");
}

// ---------------------------------------------------------------------------
// Workgroup GEMM: C[64 x NDIM] = A[64 x KDIM] @ W + epilogue
// A: row-major f16 in LDS.  W: pre-tiled f16 in global (16KB K-panels).
// Panels staged to LDS with double-buffered GLOBAL_LOAD_ASYNC_TO_LDS_B128;
// one workgroup barrier per K-step.
// B tile layout: lane l's lo 16B half at byte 16*l, hi half at 512 + 16*l,
// so each ds_load_b128 spans a contiguous 512B (minimum 2-way bank access).
// 16 waves: wave(mw,nw) -> rows mw*16..+15, cols nw*64..+63 per 256-col chunk.
// MODE 0: out f16 = acc+bias       MODE 3: out f16 = gelu(acc+bias)
// MODE 1: fused f32 = acc+bias+residual (+f16 copy)   MODE 2: fused f32 += acc+bias
// ---------------------------------------------------------------------------
template <int KDIM, int NDIM, int MODE>
__device__ __forceinline__ void gemm_tile(
    char* smemc, const _Float16* __restrict__ Wg,
    int aOff, int aStride, int outOff, int outStride,
    const float* __restrict__ bias,
    const float* __restrict__ g0, const float* __restrict__ g1,
    const float* __restrict__ g2, long rowBase, int out2Off)
{
  const int tid  = threadIdx.x;
  const int lane = tid & 31;
  const int wave = tid >> 5;
  const int mw = wave >> 2;             // 0..3
  const int nw = wave & 3;              // 0..3
  const int arow = mw * 16 + (lane & 15);
  const int ak0  = (lane >> 4) << 4;    // contiguous K-split: lane half h holds K 16h..16h+15
  constexpr int KT = KDIM / 32;
  constexpr int NC = NDIM / 256;
  const _Float16* A = (const _Float16*)(smemc + aOff);
  _Float16* panelP[2] = { (_Float16*)(smemc + OFF_BPAN),
                          (_Float16*)(smemc + OFF_BPAN + 16384) };
  const unsigned panelLds = lds_addr32(smemc + OFF_BPAN) + (unsigned)tid * 32u;
  (void)outStride; (void)g0; (void)g1; (void)g2; (void)rowBase; (void)out2Off;

  for (int nc = 0; nc < NC; ++nc) {
    v8f acc[4];
#pragma unroll
    for (int j = 0; j < 4; ++j)
#pragma unroll
      for (int e = 0; e < 8; ++e) acc[j][e] = 0.0f;

    __syncthreads();  // previous consumers of the panel buffers are done
    panel_async_copy32(panelLds, Wg + (size_t)nc * KT * 8192 + tid * 16);

    for (int kt = 0; kt < KT; ++kt) {
      wait_async0();                       // my writes to panel[kt&1] complete
      __syncthreads();                     // everyone's writes done; opposite
                                           // buffer's readers also done
      if (kt + 1 < KT)                     // overlap next panel with WMMA
        panel_async_copy32(panelLds + (unsigned)((kt + 1) & 1) * 16384u,
                           Wg + ((size_t)nc * KT + kt + 1) * 8192 + tid * 16);

      const _Float16* panel = panelP[kt & 1];
      const _Float16* ap = A + arow * aStride + kt * 32 + ak0;
      v16h a = pack16(*(const v8h*)ap, *(const v8h*)(ap + 8));
#pragma unroll
      for (int j = 0; j < 4; ++j) {
        const _Float16* bp = panel + (nw * 4 + j) * 512 + lane * 8;
        v16h b = pack16(*(const v8h*)bp, *(const v8h*)(bp + 256));
        acc[j] = __builtin_amdgcn_wmma_f32_16x16x32_f16(
            false, a, false, b, (short)0, acc[j], false, false);
      }
    }
    __syncthreads();  // all A/B reads done before epilogue may alias

    // epilogue: C layout = VGPR i, lanes 0-15 row M=i / lanes 16-31 row M=8+i
#pragma unroll
    for (int j = 0; j < 4; ++j) {
      const int col = nc * 256 + (nw * 4 + j) * 16 + (lane & 15);
#pragma unroll
      for (int i = 0; i < 8; ++i) {
        const int row = mw * 16 + i + ((lane >> 4) << 3);
        float v = acc[j][i];
        if constexpr (MODE == 0 || MODE == 3) {
          v += bias[col];
          if constexpr (MODE == 3)
            v = 0.5f * v * (1.0f + erff(v * 0.70710678118654752f));  // exact GELU
          ((_Float16*)(smemc + outOff))[row * outStride + col] = (_Float16)v;
        } else if constexpr (MODE == 1) {
          const size_t gi = (size_t)(rowBase + row) * 256 + col;
          v += bias[col] + (g0[gi] + g1[gi] + g2[gi]) * (1.0f / 3.0f);  // +residual
          ((float*)(smemc + outOff))[row * 256 + col] = v;
          ((_Float16*)(smemc + out2Off))[row * LDA + col] = (_Float16)v;
        } else {  // MODE 2: accumulate FFN output into fused f32
          float* f = (float*)(smemc + outOff) + row * 256 + col;
          *f += v + bias[col];
        }
      }
    }
  }
}

// LN of a [64 x 256] global tile -> f16 LDS. 8 lanes per row.
__device__ __forceinline__ void layernorm_tile(
    const float* __restrict__ src, long rowBase,
    const float* __restrict__ g, const float* __restrict__ b,
    char* smemc, int dstOff)
{
  const int tid = threadIdx.x;
  const int row = tid >> 3;
  const int cb  = (tid & 7) * 32;
  const float4* p = (const float4*)(src + (size_t)(rowBase + row) * 256 + cb);
  float x[32]; float s = 0.f, ss = 0.f;
#pragma unroll
  for (int i = 0; i < 8; ++i) {
    float4 t = p[i];
    x[4*i+0]=t.x; x[4*i+1]=t.y; x[4*i+2]=t.z; x[4*i+3]=t.w;
  }
#pragma unroll
  for (int i = 0; i < 32; ++i) { s += x[i]; ss += x[i] * x[i]; }
#pragma unroll
  for (int m = 4; m >= 1; m >>= 1) { s += __shfl_xor(s, m, 32); ss += __shfl_xor(ss, m, 32); }
  const float mean = s * (1.f / 256.f);
  const float var  = ss * (1.f / 256.f) - mean * mean;
  const float rs   = rsqrtf(var + 1e-5f);
  v8h* dst = (v8h*)((_Float16*)(smemc + dstOff) + row * LDA + cb);
#pragma unroll
  for (int c = 0; c < 4; ++c) {
    v8h pk;
#pragma unroll
    for (int i = 0; i < 8; ++i) {
      const int e = c * 8 + i;
      pk[i] = (_Float16)((x[e] - mean) * rs * g[cb + e] + b[cb + e]);
    }
    dst[c] = pk;
  }
}

// in-place: P0,P1,P2 = k24,k77,kx  ->  k77-kx, k24-kx, k24-k77
__device__ __forceinline__ void dk_transform(char* smemc)
{
  const int tid = threadIdx.x;
  const int row = tid >> 3;
  const int cb  = (tid & 7) * 32;
  v8h* A0 = (v8h*)((_Float16*)(smemc + OFF_P0) + row * LDA + cb);
  v8h* A1 = (v8h*)((_Float16*)(smemc + OFF_P1) + row * LDA + cb);
  v8h* A2 = (v8h*)((_Float16*)(smemc + OFF_P2) + row * LDA + cb);
#pragma unroll
  for (int c = 0; c < 4; ++c) {
    v8h a = A0[c], b = A1[c], cc = A2[c];
    v8h r0, r1, r2;
#pragma unroll
    for (int i = 0; i < 8; ++i) {
      const float fa = (float)a[i], fb = (float)b[i], fc = (float)cc[i];
      r0[i] = (_Float16)(fb - fc);
      r1[i] = (_Float16)(fa - fc);
      r2[i] = (_Float16)(fa - fb);
    }
    A0[c] = r0; A1[c] = r1; A2[c] = r2;
  }
}

// d_s[row][h][g] = scale * q[row,32h..] . dk[row,32g..]
__device__ __forceinline__ void score_tile(char* smemc, int dkOff, int sIdx)
{
  const int tid = threadIdx.x;
  const int row = tid >> 3;
  const int h   = tid & 7;
  const v8h* q8 = (const v8h*)((const _Float16*)(smemc + OFF_P3) + row * LDA + h * 32);
  float qv[32];
#pragma unroll
  for (int c = 0; c < 4; ++c) {
    v8h t = q8[c];
#pragma unroll
    for (int i = 0; i < 8; ++i) qv[c * 8 + i] = (float)t[i];
  }
  float* dd = (float*)(smemc + OFF_D) + row * 192 + sIdx * 64 + h * 8;
  const _Float16* kb = (const _Float16*)(smemc + dkOff) + row * LDA;
  for (int gg = 0; gg < 8; ++gg) {
    const v8h* k8 = (const v8h*)(kb + gg * 32);
    float acc = 0.f;
#pragma unroll
    for (int c = 0; c < 4; ++c) {
      v8h t = k8[c];
#pragma unroll
      for (int i = 0; i < 8; ++i) acc += qv[c * 8 + i] * (float)t[i];
    }
    dd[gg] = acc * 0.17677669529663687f;  // 32^-0.5
  }
}

// mixed[row][32h+d] = rw0*out24 + rw1*out77 + rw2*outx  -> f16 at P3
__device__ __forceinline__ void mix_tile(char* smemc, const float* __restrict__ radar)
{
  const int tid = threadIdx.x;
  const int row = tid >> 3;
  const int h   = tid & 7;
  const float r0 = radar[0], r1 = radar[1], r2 = radar[2];
  const float mx = fmaxf(r0, fmaxf(r1, r2));
  const float e0 = __expf(r0 - mx), e1 = __expf(r1 - mx), e2 = __expf(r2 - mx);
  const float inv = 1.f / (e0 + e1 + e2);
  const float rw0 = e0 * inv, rw1 = e1 * inv, rw2 = e2 * inv;
  const float* dd = (const float*)(smemc + OFF_D) + row * 192;
  const _Float16* v24 = (const _Float16*)(smemc + OFF_P0) + row * LDA;
  const _Float16* v77 = (const _Float16*)(smemc + OFF_P1) + row * LDA;
  const _Float16* vx  = (const _Float16*)(smemc + OFF_P2) + row * LDA;
  float o[32];
#pragma unroll
  for (int d = 0; d < 32; ++d) o[d] = 0.f;
  for (int gg = 0; gg < 8; ++gg) {
    const float p24 = 1.f / (1.f + __expf(-dd[       h * 8 + gg]));  // pair-softmax == sigmoid
    const float p77 = 1.f / (1.f + __expf(-dd[ 64 +  h * 8 + gg]));
    const float px  = 1.f / (1.f + __expf(-dd[128 +  h * 8 + gg]));
    const v8h* a24 = (const v8h*)(v24 + gg * 32);
    const v8h* a77 = (const v8h*)(v77 + gg * 32);
    const v8h* ax  = (const v8h*)(vx  + gg * 32);
#pragma unroll
    for (int c = 0; c < 4; ++c) {
      v8h t24 = a24[c], t77 = a77[c], tx = ax[c];
#pragma unroll
      for (int i = 0; i < 8; ++i) {
        const int d = c * 8 + i;
        const float w24 = (float)t24[i], w77 = (float)t77[i], wx = (float)tx[i];
        o[d] += rw0 * (p24 * w77 + (1.f - p24) * wx)
              + rw1 * (p77 * w24 + (1.f - p77) * wx)
              + rw2 * (px  * w24 + (1.f - px ) * w77);
      }
    }
  }
  v8h* dst = (v8h*)((_Float16*)(smemc + OFF_P3) + row * LDA + h * 32);
#pragma unroll
  for (int c = 0; c < 4; ++c) {
    v8h pk;
#pragma unroll
    for (int i = 0; i < 8; ++i) pk[i] = (_Float16)o[c * 8 + i];
    dst[c] = pk;
  }
}

__device__ __forceinline__ void final_ln(char* smemc, long rowBase,
    const float* __restrict__ g, const float* __restrict__ b, float* __restrict__ out)
{
  const int tid = threadIdx.x;
  const int row = tid >> 3;
  const int cb  = (tid & 7) * 32;
  const float* src = (const float*)(smemc + OFF_F32) + row * 256 + cb;
  float x[32]; float s = 0.f, ss = 0.f;
#pragma unroll
  for (int i = 0; i < 32; ++i) { x[i] = src[i]; s += x[i]; ss += x[i] * x[i]; }
#pragma unroll
  for (int m = 4; m >= 1; m >>= 1) { s += __shfl_xor(s, m, 32); ss += __shfl_xor(ss, m, 32); }
  const float mean = s * (1.f / 256.f);
  const float var  = ss * (1.f / 256.f) - mean * mean;
  const float rs   = rsqrtf(var + 1e-5f);
  float4* op = (float4*)(out + (size_t)(rowBase + row) * 256 + cb);
#pragma unroll
  for (int c = 0; c < 8; ++c) {
    float4 t;
    t.x = (x[4*c+0] - mean) * rs * g[cb + 4*c+0] + b[cb + 4*c+0];
    t.y = (x[4*c+1] - mean) * rs * g[cb + 4*c+1] + b[cb + 4*c+1];
    t.z = (x[4*c+2] - mean) * rs * g[cb + 4*c+2] + b[cb + 4*c+2];
    t.w = (x[4*c+3] - mean) * rs * g[cb + 4*c+3] + b[cb + 4*c+3];
    op[c] = t;
  }
}

// ---------------------------------------------------------------------------
__global__ void __launch_bounds__(NTH, 1)
fused_attn_kernel(const float* f24, const float* f77, const float* fx,
                  const float* norm_g, const float* norm_b,
                  const _Float16* wsW,
                  const float* bq24, const float* bk24, const float* bv24,
                  const float* bq77, const float* bk77, const float* bv77,
                  const float* bqx,  const float* bkx,  const float* bvx,
                  const float* bout, const float* b1, const float* b2,
                  const float* radar, const float* fnorm_g, const float* fnorm_b,
                  float* out, long Btot)
{
  extern __shared__ char smemc[];
  const long rowBase = (long)blockIdx.x * MT;

  layernorm_tile(f24, rowBase, norm_g, norm_b, smemc, OFF_N24);
  layernorm_tile(f77, rowBase, norm_g, norm_b, smemc, OFF_N77);
  layernorm_tile(fx,  rowBase, norm_g, norm_b, smemc, OFF_NX);
  __syncthreads();

  // K projections -> P0,P1,P2
  gemm_tile<256,256,0>(smemc, wsW + WK24, OFF_N24, LDA, OFF_P0, LDA, bk24, nullptr,nullptr,nullptr,0,0);
  gemm_tile<256,256,0>(smemc, wsW + WK77, OFF_N77, LDA, OFF_P1, LDA, bk77, nullptr,nullptr,nullptr,0,0);
  gemm_tile<256,256,0>(smemc, wsW + WKX,  OFF_NX,  LDA, OFF_P2, LDA, bkx,  nullptr,nullptr,nullptr,0,0);
  __syncthreads();
  dk_transform(smemc);
  __syncthreads();

  // Q projections + diff-scores
  gemm_tile<256,256,0>(smemc, wsW + WQ24, OFF_N24, LDA, OFF_P3, LDA, bq24, nullptr,nullptr,nullptr,0,0);
  __syncthreads();
  score_tile(smemc, OFF_P0, 0);
  __syncthreads();
  gemm_tile<256,256,0>(smemc, wsW + WQ77, OFF_N77, LDA, OFF_P3, LDA, bq77, nullptr,nullptr,nullptr,0,0);
  __syncthreads();
  score_tile(smemc, OFF_P1, 1);
  __syncthreads();
  gemm_tile<256,256,0>(smemc, wsW + WQX, OFF_NX, LDA, OFF_P3, LDA, bqx, nullptr,nullptr,nullptr,0,0);
  __syncthreads();
  score_tile(smemc, OFF_P2, 2);
  __syncthreads();

  // V projections -> P0,P1,P2 (dk buffers dead)
  gemm_tile<256,256,0>(smemc, wsW + WV24, OFF_N24, LDA, OFF_P0, LDA, bv24, nullptr,nullptr,nullptr,0,0);
  gemm_tile<256,256,0>(smemc, wsW + WV77, OFF_N77, LDA, OFF_P1, LDA, bv77, nullptr,nullptr,nullptr,0,0);
  gemm_tile<256,256,0>(smemc, wsW + WVX,  OFF_NX,  LDA, OFF_P2, LDA, bvx,  nullptr,nullptr,nullptr,0,0);
  __syncthreads();
  mix_tile(smemc, radar);           // -> mixed f16 at P3
  __syncthreads();

  // Wout + bias + residual -> fused f32 (P1 region) + f16 copy (P3)
  gemm_tile<256,256,1>(smemc, wsW + WOUT, OFF_P3, LDA, OFF_F32, 0, bout, f24, f77, fx, rowBase, OFF_P3);
  __syncthreads();
  // FFN1: gelu(fused @ W1 + b1) -> h1 f16 at offset 0 (n/P0 regions dead)
  gemm_tile<256,1024,3>(smemc, wsW + WFF1, OFF_P3, LDA, 0, LDH1, b1, nullptr,nullptr,nullptr,0,0);
  __syncthreads();
  // FFN2: fused f32 += h1 @ W2 + b2
  gemm_tile<1024,256,2>(smemc, wsW + WFF2, 0, LDH1, OFF_F32, 0, b2, nullptr,nullptr,nullptr,0,0);
  __syncthreads();

  final_ln(smemc, rowBase, fnorm_g, fnorm_b, out);

  if (blockIdx.x == 0 && threadIdx.x < 3) {  // second output: rw = softmax(radar_w)
    const float r0 = radar[0], r1 = radar[1], r2 = radar[2];
    const float mx = fmaxf(r0, fmaxf(r1, r2));
    const float e0 = __expf(r0 - mx), e1 = __expf(r1 - mx), e2 = __expf(r2 - mx);
    const float inv = 1.f / (e0 + e1 + e2);
    const float rwv = (threadIdx.x == 0 ? e0 : threadIdx.x == 1 ? e1 : e2) * inv;
    out[(size_t)Btot * 256 + threadIdx.x] = rwv;
  }
}

// f32 [K][N] row-major -> f16 WMMA-B tiled: 32x16 tiles, tiles grouped so each
// (nc, kt) K-panel (16 tiles = 16KB) is contiguous. Within a tile, lane l's
// lo 16B half (K k0..k0+7) is at element l*8 and its hi half (K k0+8..k0+15)
// at 256 + l*8, so B-fragment ds_load_b128s span contiguous 512B (min bank access).
__global__ void prep_weight_kernel(const float* __restrict__ W, _Float16* __restrict__ dst,
                                   int K, int N)
{
  const int idx = blockIdx.x * blockDim.x + threadIdx.x;
  if (idx >= K * N) return;
  const int k = idx / N;
  const int n = idx - k * N;
  const int nc   = n >> 8;
  const int ntc  = (n >> 4) & 15;
  const int kt   = k >> 5;
  const int lane = (((k >> 4) & 1) << 4) | (n & 15);
  const int half = (k >> 3) & 1;
  const int tile = (nc * (K >> 5) + kt) * 16 + ntc;
  dst[(size_t)tile * 512 + half * 256 + lane * 8 + (k & 7)] = (_Float16)W[idx];
}

extern "C" void kernel_launch(void* const* d_in, const int* in_sizes, int n_in,
                              void* d_out, int out_size, void* d_ws, size_t ws_size,
                              hipStream_t stream)
{
  (void)n_in; (void)out_size; (void)ws_size;
  _Float16* wsh = (_Float16*)d_ws;

  struct WMap { int idx, K, N; size_t off; };
  const WMap wm[12] = {
    {5,256,256,WQ24},{7,256,256,WK24},{9,256,256,WV24},
    {11,256,256,WQ77},{13,256,256,WK77},{15,256,256,WV77},
    {17,256,256,WQX},{19,256,256,WKX},{21,256,256,WVX},
    {23,256,256,WOUT},{26,256,1024,WFF1},{28,1024,256,WFF2}};
  for (int i = 0; i < 12; ++i) {
    const int total = wm[i].K * wm[i].N;
    prep_weight_kernel<<<(total + 255) / 256, 256, 0, stream>>>(
        (const float*)d_in[wm[i].idx], wsh + wm[i].off, wm[i].K, wm[i].N);
  }

  const long B = (long)in_sizes[0] / 256;
  hipFuncSetAttribute((const void*)fused_attn_kernel,
                      hipFuncAttributeMaxDynamicSharedMemorySize, SMEM_BYTES);
  fused_attn_kernel<<<(int)(B / MT), NTH, SMEM_BYTES, stream>>>(
      (const float*)d_in[0], (const float*)d_in[1], (const float*)d_in[2],
      (const float*)d_in[3], (const float*)d_in[4],
      wsh,
      (const float*)d_in[6],  (const float*)d_in[8],  (const float*)d_in[10],
      (const float*)d_in[12], (const float*)d_in[14], (const float*)d_in[16],
      (const float*)d_in[18], (const float*)d_in[20], (const float*)d_in[22],
      (const float*)d_in[24], (const float*)d_in[27], (const float*)d_in[29],
      (const float*)d_in[25], (const float*)d_in[30], (const float*)d_in[31],
      (float*)d_out, B);
}
